// MultiHeadAttention_21577915695389
// MI455X (gfx1250) — compile-verified, matrix operand order checked by reading, not run on hardware
//
#include <hip/hip_runtime.h>

typedef unsigned short u16;
typedef __attribute__((ext_vector_type(16))) __bf16 v16bf;
typedef __attribute__((ext_vector_type(8)))  __bf16 v8bf;
typedef __attribute__((ext_vector_type(8)))  float  v8f;
typedef int v4i __attribute__((vector_size(16)));

#define D_MODEL 1024
#define NHEADS  16
#define DK      64
#define SEQ     2048
#define BATCH   2
#define MROWS   (BATCH * SEQ)      // 4096 rows for all big GEMMs

// LDS row strides (in u16 elements), padded for conflict-free b128 reads:
// 16 lanes read rows r..r+15; stride*2B mod 256B must visit distinct banks.
#define SB_STRIDE 40   // 32-wide k tiles   (80B  = 20 banks/row)
#define SK_STRIDE 88   // 64-wide tiles     (176B = 44 banks/row)
#define SP_STRIDE 72   // P staging 64 wide (144B = 36 banks/row)

// ---------------- bf16 split helpers (round-to-nearest-even) ----------------
__device__ __forceinline__ u16 f2bf(float x) {
  unsigned u = __float_as_uint(x);
  u += 0x7FFFu + ((u >> 16) & 1u);
  return (u16)(u >> 16);
}
__device__ __forceinline__ float bf2f(u16 h) {
  return __uint_as_float(((unsigned)h) << 16);
}

// ------------- WMMA operand load: 16-bit A/B matrix, 16x32 tile -------------
__device__ __forceinline__ v16bf load_op16(const u16* base, int stride,
                                           int row, int k0, int half) {
  const u16* p = base + (size_t)row * (size_t)stride + k0 + half * 8;
  v8bf c0 = *(const v8bf*)(p);
  v8bf c1 = *(const v8bf*)(p + 16);
  return __builtin_shufflevector(c0, c1, 0, 1, 2, 3, 4, 5, 6, 7,
                                 8, 9, 10, 11, 12, 13, 14, 15);
}

__device__ __forceinline__ v8f wmma_bf16(v16bf a, v16bf b, v8f c) {
  return __builtin_amdgcn_wmma_f32_16x16x32_bf16(false, a, false, b,
                                                 (short)0, c, false, false);
}

// ---------------- async global -> LDS copy (16B per lane) -------------------
typedef __attribute__((address_space(1))) v4i* gvec_t;
typedef __attribute__((address_space(3))) v4i* lvec_t;

__device__ __forceinline__ void async_copy16(const u16* g, u16* l) {
  gvec_t gp = (gvec_t)g;
  lvec_t lp = (lvec_t)l;
#if __has_builtin(__builtin_amdgcn_global_load_async_to_lds_b128)
  __builtin_amdgcn_global_load_async_to_lds_b128(gp, lp, 0, 0);
#else
  asm volatile("global_load_async_to_lds_b128 %0, %1, off"
               :: "v"(lp), "v"(gp) : "memory");
#endif
}

template <int N>
__device__ __forceinline__ void wait_async() {
#if __has_builtin(__builtin_amdgcn_s_wait_asynccnt)
  __builtin_amdgcn_s_wait_asynccnt((unsigned short)N);
#else
  asm volatile("s_wait_asynccnt %0" :: "i"(N) : "memory");
#endif
}

// ---------------- prep: fp32 -> bf16 hi/lo split (elementwise) --------------
__global__ void k_split(const float* __restrict__ x, u16* __restrict__ hi,
                        u16* __restrict__ lo, int n) {
  int i = blockIdx.x * blockDim.x + threadIdx.x;
  if (i < n) {
    float v = x[i];
    u16 h = f2bf(v);
    hi[i] = h;
    lo[i] = f2bf(v - bf2f(h));
  }
}

// ------- prep: W[k][n] fp32 -> Wt hi/lo bf16 [n][k] (B-operand layout) ------
__global__ void k_wt(const float* __restrict__ w, u16* __restrict__ hi,
                     u16* __restrict__ lo) {
  int i = blockIdx.x * blockDim.x + threadIdx.x;
  if (i < D_MODEL * D_MODEL) {
    int k = i >> 10, n = i & 1023;
    float v = w[i];
    u16 h = f2bf(v);
    size_t o = (size_t)n * D_MODEL + k;
    hi[o] = h;
    lo[o] = f2bf(v - bf2f(h));
  }
}

// ---------------------------- bf16x3 GEMM kernel ----------------------------
// Workgroup = 8 waves stacked in M: 128x64 output block.  B tiles (64n x 32k)
// are async-staged into LDS once per WG (8x traffic cut), double buffered.
// mode 0: split bf16 -> [B,H,S,64] (+ optional k_pe add)
// mode 1: split bf16 -> [B,H,64,S]
// mode 2: fp32 -> Of[m][n]
__global__ __launch_bounds__(256) void k_gemm(
    const u16* __restrict__ Ah, const u16* __restrict__ Al,
    const u16* __restrict__ Bh, const u16* __restrict__ Bl,
    const float* __restrict__ bias, const float* __restrict__ kpe,
    u16* __restrict__ Oh, u16* __restrict__ Ol, float* __restrict__ Of,
    int mode, float scale) {
  __shared__ __align__(16) u16 sB[2][2][64 * SB_STRIDE];

  const int tid = threadIdx.x;
  const int lane = tid & 31;
  const int w = tid >> 5;
  const int m0 = (blockIdx.x >> 4) * 128 + w * 16;
  const int n0 = (blockIdx.x & 15) * 64;
  const int half = lane >> 4, cl = lane & 15;

  // staging map: thread -> one 16B chunk per plane
  const int sn = tid >> 2;            // 0..63  (row n_rel)
  const int sk = (tid & 3) * 8;       // 0,8,16,24 (k element offset)

  v8f acc[4];
#pragma unroll
  for (int t = 0; t < 4; ++t)
    acc[t] = (v8f){0.f, 0.f, 0.f, 0.f, 0.f, 0.f, 0.f, 0.f};

  // prologue: stage k-block 0 into buffer 0
  {
    const size_t go = (size_t)(n0 + sn) * D_MODEL + sk;
    async_copy16(Bh + go, &sB[0][0][sn * SB_STRIDE + sk]);
    async_copy16(Bl + go, &sB[0][1][sn * SB_STRIDE + sk]);
  }

#pragma unroll 1
  for (int i = 0; i < D_MODEL / 32; ++i) {
    const int p = i & 1;
    const int kb = i * 32;
    if (i + 1 < D_MODEL / 32) {
      const size_t go = (size_t)(n0 + sn) * D_MODEL + (kb + 32) + sk;
      async_copy16(Bh + go, &sB[1 - p][0][sn * SB_STRIDE + sk]);
      async_copy16(Bl + go, &sB[1 - p][1][sn * SB_STRIDE + sk]);
      wait_async<2>();
    } else {
      wait_async<0>();
    }
    __syncthreads();   // buffer p fully staged by all waves

    __builtin_prefetch(Ah + (size_t)(m0 + cl) * D_MODEL + kb + 64, 0, 0);
    v16bf a_h = load_op16(Ah, D_MODEL, m0 + cl, kb, half);
    v16bf a_l = load_op16(Al, D_MODEL, m0 + cl, kb, half);
#pragma unroll
    for (int t = 0; t < 4; ++t) {
      v16bf b_h = load_op16(&sB[p][0][0], SB_STRIDE, t * 16 + cl, 0, half);
      v16bf b_l = load_op16(&sB[p][1][0], SB_STRIDE, t * 16 + cl, 0, half);
      acc[t] = wmma_bf16(a_h, b_h, acc[t]);
      acc[t] = wmma_bf16(a_h, b_l, acc[t]);
      acc[t] = wmma_bf16(a_l, b_h, acc[t]);
    }
    __syncthreads();   // all reads of buffer p done before the next iteration re-stages it
  }

  // epilogue.  C layout: lane holds col n0+t*16+cl; VGPR r -> row r + 8*half.
#pragma unroll
  for (int t = 0; t < 4; ++t) {
    int n = n0 + t * 16 + cl;
    float bn = bias[n];
#pragma unroll
    for (int r = 0; r < 8; ++r) {
      int m = m0 + r + 8 * half;
      float val = (acc[t][r] + bn) * scale;
      if (mode == 2) {
        Of[(size_t)m * D_MODEL + n] = val;
      } else {
        int b = m >> 11, s = m & (SEQ - 1);
        int h = n >> 6, d = n & (DK - 1);
        if (kpe) val += kpe[(size_t)(h * DK + d) * SEQ + s];
        size_t idx = (mode == 0)
                         ? ((size_t)((b * NHEADS + h) * SEQ + s)) * DK + d
                         : ((size_t)((b * NHEADS + h) * DK + d)) * SEQ + s;
        u16 hb = f2bf(val);
        Oh[idx] = hb;
        Ol[idx] = f2bf(val - bf2f(hb));
      }
    }
  }
}

// ------------------------ flash attention (bf16x3) --------------------------
// 8 waves of a WG share one (b,h); K/V 64-key tiles async-staged to LDS once
// per WG (8x traffic cut), double buffered.  Per wave: 16 query rows,
// online softmax, P staged through per-wave LDS into A-operand layout.
__global__ __launch_bounds__(256) void k_attn(
    const u16* __restrict__ Qh, const u16* __restrict__ Ql,
    const u16* __restrict__ Kh, const u16* __restrict__ Kl,
    const u16* __restrict__ Vh, const u16* __restrict__ Vl,
    u16* __restrict__ Oh, u16* __restrict__ Ol) {
  __shared__ __align__(16) u16 sK[2][2][64 * SK_STRIDE];
  __shared__ __align__(16) u16 sV[2][2][64 * SK_STRIDE];
  __shared__ __align__(16) u16 sP[8][2][16 * SP_STRIDE];

  const int tid = threadIdx.x;
  const int lane = tid & 31;
  const int w = tid >> 5;
  const int wid = blockIdx.x * 8 + w;
  const int bh = wid >> 7;          // b*16 + h (all 8 waves of a WG agree)
  const int m0 = (wid & 127) * 16;
  const int half = lane >> 4, cl = lane & 15;

  const u16* qbh = Qh + (size_t)bh * SEQ * DK;
  const u16* qbl = Ql + (size_t)bh * SEQ * DK;
  const u16* kgh = Kh + (size_t)bh * SEQ * DK;   // [t][d]
  const u16* kgl = Kl + (size_t)bh * SEQ * DK;
  const u16* vgh = Vh + (size_t)bh * DK * SEQ;   // [d][t]
  const u16* vgl = Vl + (size_t)bh * DK * SEQ;

  // staging map: per plane, thread covers chunks (row, off) and (row+32, off)
  const int srow = tid >> 3;          // 0..31
  const int soff = (tid & 7) * 8;     // element offset 0..56

  v16bf q_h[2], q_l[2];
#pragma unroll
  for (int kk = 0; kk < 2; ++kk) {
    q_h[kk] = load_op16(qbh, DK, m0 + cl, kk * 32, half);
    q_l[kk] = load_op16(qbl, DK, m0 + cl, kk * 32, half);
  }

  float mr[8], lr[8];
  v8f o[4];
#pragma unroll
  for (int r = 0; r < 8; ++r) { mr[r] = -3.0e38f; lr[r] = 0.f; }
#pragma unroll
  for (int t = 0; t < 4; ++t)
    o[t] = (v8f){0.f, 0.f, 0.f, 0.f, 0.f, 0.f, 0.f, 0.f};

  // stage one 64-key tile (K and V, hi+lo) into LDS buffer `buf`
  auto stage = [&](int buf, int t0) {
#pragma unroll
    for (int j = 0; j < 2; ++j) {
      int row = srow + j * 32;
      async_copy16(kgh + (size_t)(t0 + row) * DK + soff,
                   &sK[buf][0][row * SK_STRIDE + soff]);
      async_copy16(kgl + (size_t)(t0 + row) * DK + soff,
                   &sK[buf][1][row * SK_STRIDE + soff]);
      async_copy16(vgh + (size_t)row * SEQ + t0 + soff,
                   &sV[buf][0][row * SK_STRIDE + soff]);
      async_copy16(vgl + (size_t)row * SEQ + t0 + soff,
                   &sV[buf][1][row * SK_STRIDE + soff]);
    }
  };

  stage(0, 0);

#pragma unroll 1
  for (int i = 0; i < SEQ / 64; ++i) {
    const int p = i & 1;
    const int t0 = i * 64;
    if (i + 1 < SEQ / 64) { stage(1 - p, t0 + 64); wait_async<8>(); }
    else                  { wait_async<0>(); }
    __syncthreads();   // K/V buffer p fully staged

    // ---- scores S = Qs * Keff^T  (16 x 64 block, bf16x3) ----
    v8f sc[4];
#pragma unroll
    for (int st = 0; st < 4; ++st) {
      v8f c = (v8f){0.f, 0.f, 0.f, 0.f, 0.f, 0.f, 0.f, 0.f};
#pragma unroll
      for (int kk = 0; kk < 2; ++kk) {
        v16bf b_h = load_op16(&sK[p][0][0], SK_STRIDE, st * 16 + cl, kk * 32, half);
        v16bf b_l = load_op16(&sK[p][1][0], SK_STRIDE, st * 16 + cl, kk * 32, half);
        c = wmma_bf16(q_h[kk], b_h, c);
        c = wmma_bf16(q_h[kk], b_l, c);
        c = wmma_bf16(q_l[kk], b_h, c);
      }
      sc[st] = c;
    }
    // ---- block row-max (cross-lane within 16-lane half) ----
    float bm[8];
#pragma unroll
    for (int r = 0; r < 8; ++r)
      bm[r] = fmaxf(fmaxf(sc[0][r], sc[1][r]), fmaxf(sc[2][r], sc[3][r]));
#pragma unroll
    for (int r = 0; r < 8; ++r) {
      bm[r] = fmaxf(bm[r], __shfl_xor(bm[r], 1, 16));
      bm[r] = fmaxf(bm[r], __shfl_xor(bm[r], 2, 16));
      bm[r] = fmaxf(bm[r], __shfl_xor(bm[r], 4, 16));
      bm[r] = fmaxf(bm[r], __shfl_xor(bm[r], 8, 16));
    }
    float al[8], rs[8];
#pragma unroll
    for (int r = 0; r < 8; ++r) {
      float mn = fmaxf(mr[r], bm[r]);
      al[r] = __expf(mr[r] - mn);
      mr[r] = mn;
      rs[r] = 0.f;
    }
    // ---- P = exp(S - m), split hi/lo into per-wave LDS ----
#pragma unroll
    for (int st = 0; st < 4; ++st) {
#pragma unroll
      for (int r = 0; r < 8; ++r) {
        float pv = __expf(sc[st][r] - mr[r]);
        rs[r] += pv;
        u16 hb = f2bf(pv);
        int off = (r + 8 * half) * SP_STRIDE + st * 16 + cl;
        sP[w][0][off] = hb;
        sP[w][1][off] = f2bf(pv - bf2f(hb));
      }
    }
#pragma unroll
    for (int r = 0; r < 8; ++r) {
      rs[r] += __shfl_xor(rs[r], 1, 16);
      rs[r] += __shfl_xor(rs[r], 2, 16);
      rs[r] += __shfl_xor(rs[r], 4, 16);
      rs[r] += __shfl_xor(rs[r], 8, 16);
      lr[r] = lr[r] * al[r] + rs[r];
    }
    v8f av = (v8f){al[0], al[1], al[2], al[3], al[4], al[5], al[6], al[7]};
#pragma unroll
    for (int t = 0; t < 4; ++t) o[t] *= av;

    asm volatile("s_wait_dscnt 0" ::: "memory");  // intra-wave P store->load

    v16bf p_h[2], p_l[2];
#pragma unroll
    for (int kk = 0; kk < 2; ++kk) {
      p_h[kk] = load_op16(&sP[w][0][0], SP_STRIDE, cl, kk * 32, half);
      p_l[kk] = load_op16(&sP[w][1][0], SP_STRIDE, cl, kk * 32, half);
    }
    // ---- O += P * V  (V^T tile in LDS) ----
#pragma unroll
    for (int dt = 0; dt < 4; ++dt) {
#pragma unroll
      for (int kk = 0; kk < 2; ++kk) {
        v16bf v_h = load_op16(&sV[p][0][0], SK_STRIDE, dt * 16 + cl, kk * 32, half);
        v16bf v_l = load_op16(&sV[p][1][0], SK_STRIDE, dt * 16 + cl, kk * 32, half);
        o[dt] = wmma_bf16(p_h[kk], v_h, o[dt]);
        o[dt] = wmma_bf16(p_h[kk], v_l, o[dt]);
        o[dt] = wmma_bf16(p_l[kk], v_h, o[dt]);
      }
    }
    __syncthreads();   // everyone done with K/V buffer p before re-staging
  }

  // ---- normalize, split, store as [B*S][1024] bf16 hi/lo ----
  const int b = bh >> 4, h = bh & 15;
#pragma unroll
  for (int t = 0; t < 4; ++t) {
#pragma unroll
    for (int r = 0; r < 8; ++r) {
      float val = o[t][r] / lr[r];
      int grow = b * SEQ + m0 + r + 8 * half;
      int col = h * DK + t * 16 + cl;
      size_t idx = (size_t)grow * D_MODEL + col;
      u16 hb = f2bf(val);
      Oh[idx] = hb;
      Ol[idx] = f2bf(val - bf2f(hb));
    }
  }
}

// ------------------------------- launcher -----------------------------------
extern "C" void kernel_launch(void* const* d_in, const int* in_sizes, int n_in,
                              void* d_out, int out_size, void* d_ws,
                              size_t ws_size, hipStream_t stream) {
  const float* q   = (const float*)d_in[0];
  const float* k   = (const float*)d_in[1];
  const float* v   = (const float*)d_in[2];
  const float* kpe = (const float*)d_in[3];
  const float* Wq  = (const float*)d_in[4];
  const float* bq  = (const float*)d_in[5];
  const float* Wk  = (const float*)d_in[6];
  const float* bk  = (const float*)d_in[7];
  const float* Wv  = (const float*)d_in[8];
  const float* bv  = (const float*)d_in[9];
  const float* Wo  = (const float*)d_in[10];
  const float* bo  = (const float*)d_in[11];
  float* out = (float*)d_out;

  const size_t NX = (size_t)MROWS * D_MODEL;      // 4M elems
  const size_t NW = (size_t)D_MODEL * D_MODEL;    // 1M elems
  char* p = (char*)d_ws;
  auto take = [&](size_t elems) { u16* r = (u16*)p; p += elems * sizeof(u16); return r; };
  u16 *qs_h = take(NX), *qs_l = take(NX);
  u16 *ks_h = take(NX), *ks_l = take(NX);
  u16 *vs_h = take(NX), *vs_l = take(NX);
  u16 *wq_h = take(NW), *wq_l = take(NW);
  u16 *wk_h = take(NW), *wk_l = take(NW);
  u16 *wv_h = take(NW), *wv_l = take(NW);
  u16 *wo_h = take(NW), *wo_l = take(NW);
  u16 *qh_h = take(NX), *qh_l = take(NX);   // [B,H,S,64]   (Q, pre-scaled)
  u16 *ke_h = take(NX), *ke_l = take(NX);   // [B,H,S,64]   (kh + k_pe^T)
  u16 *vt_h = take(NX), *vt_l = take(NX);   // [B,H,64,S]   (V transposed)
  u16 *ao_h = take(NX), *ao_l = take(NX);   // [B*S,1024]   (attention out)

  dim3 blk(256);
  k_split<<<(int)((NX + 255) / 256), blk, 0, stream>>>(q, qs_h, qs_l, (int)NX);
  k_split<<<(int)((NX + 255) / 256), blk, 0, stream>>>(k, ks_h, ks_l, (int)NX);
  k_split<<<(int)((NX + 255) / 256), blk, 0, stream>>>(v, vs_h, vs_l, (int)NX);
  k_wt<<<(int)((NW + 255) / 256), blk, 0, stream>>>(Wq, wq_h, wq_l);
  k_wt<<<(int)((NW + 255) / 256), blk, 0, stream>>>(Wk, wk_h, wk_l);
  k_wt<<<(int)((NW + 255) / 256), blk, 0, stream>>>(Wv, wv_h, wv_l);
  k_wt<<<(int)((NW + 255) / 256), blk, 0, stream>>>(Wo, wo_h, wo_l);

  k_gemm<<<512, blk, 0, stream>>>(qs_h, qs_l, wq_h, wq_l, bq, nullptr,
                                  qh_h, qh_l, nullptr, 0, 0.125f);  // 1/sqrt(64)
  k_gemm<<<512, blk, 0, stream>>>(ks_h, ks_l, wk_h, wk_l, bk, kpe,
                                  ke_h, ke_l, nullptr, 0, 1.0f);
  k_gemm<<<512, blk, 0, stream>>>(vs_h, vs_l, wv_h, wv_l, bv, nullptr,
                                  vt_h, vt_l, nullptr, 1, 1.0f);
  k_attn<<<512, blk, 0, stream>>>(qh_h, qh_l, ke_h, ke_l, vt_h, vt_l,
                                  ao_h, ao_l);
  k_gemm<<<512, blk, 0, stream>>>(ao_h, ao_l, wo_h, wo_l, bo, nullptr,
                                  nullptr, nullptr, out, 2, 1.0f);
}
